// Attention_28501402976721
// MI455X (gfx1250) — matrix operand order repaired and checked by reading, not compiled, not run on hardware
//
#include <hip/hip_runtime.h>

#define BB 4
#define SS 2048
#define DD 1024
#define HH 16
#define HDIM 64

typedef __attribute__((ext_vector_type(4)))  _Float16     half4_t;
typedef __attribute__((ext_vector_type(8)))  _Float16     half8_t;
typedef __attribute__((ext_vector_type(16))) _Float16     half16_t;
typedef __attribute__((ext_vector_type(4)))  float        float4_t;
typedef __attribute__((ext_vector_type(8)))  float        float8_t;
typedef __attribute__((ext_vector_type(4)))  unsigned int vu4;
typedef __attribute__((ext_vector_type(4)))  int          vi4;
typedef __attribute__((ext_vector_type(8)))  int          vi8;

#if defined(__has_builtin)
#if __has_builtin(__builtin_amdgcn_tensor_load_to_lds) && __has_builtin(__builtin_amdgcn_s_wait_tensorcnt)
#define HAVE_TDM 1
#endif
#endif
#ifndef HAVE_TDM
#define HAVE_TDM 0
#endif

static __device__ __forceinline__ float8_t wmma_f32_16x16x32_f16(half16_t a, half16_t b, float8_t c) {
  // 8 args: (neg_a, A, neg_b, B, c_mod, C, reuse_a, reuse_b)
  return __builtin_amdgcn_wmma_f32_16x16x32_f16(false, a, false, b, (short)0, c, false, false);
}

static __device__ __forceinline__ half16_t cat16(half8_t lo, half8_t hi) {
  return __builtin_shufflevector(lo, hi, 0, 1, 2, 3, 4, 5, 6, 7, 8, 9, 10, 11, 12, 13, 14, 15);
}

// ---------------------------------------------------------------------------
// GEMM tiling: block = 256 threads = 8 waves; block tile 256(M) x 64(N), K=32.
// Each wave computes 32 rows x 64 cols: 2 A-frags, 4 B-frags, 8 WMMAs issued
// back-to-back after all ds_load_b128s. LDS row stride 40 halves (80B) =
// 16 DWORDs data + 4 DWORDs pad, matching the TDM pad encoding in out_proj.
// ---------------------------------------------------------------------------
#define GM_BM 256
#define GM_BN 64
#define GM_BK 32
#define LDS_STRIDE 40

#if HAVE_TDM
// TDM descriptor per CDNA5 ISA: 2D f16 tile
// (tile_dim0=32 elems/row, tile_dim1=256 rows) from a [8192,1024] f16 tensor,
// padded in LDS: 16 DWORDs (code 3) then 4 DWORDs pad (code 3) -> stride 40h.
static __device__ __forceinline__ void tdm_load_a_tile(const _Float16* gptr, unsigned int lds_byte_addr) {
  const unsigned long long ga = (unsigned long long)(size_t)gptr;
  vu4 g0;
  g0[0] = 1u;                                        // count=1, user descriptor
  g0[1] = lds_byte_addr;                             // lds_addr
  g0[2] = (unsigned int)ga;                          // global_addr[31:0]
  g0[3] = (unsigned int)(ga >> 32) | (2u << 30);     // global_addr[56:32] | type=2
  vi8 g1;
  g1[0] = (int)((1u << 16) |                         // data_size = 1 (2 bytes)
                (1u << 20) |                         // pad_enable
                (3u << 22) |                         // pad_interval: 16 DWORDs
                (3u << 25));                         // pad_amount: 4 DWORDs
  g1[1] = (int)(1024u << 16);                        // tensor_dim0 = 1024 (lo16)
  g1[2] = (int)(8192u << 16);                        // dim0 hi16=0 | tensor_dim1 lo16 = 8192
  g1[3] = (int)(32u << 16);                          // dim1 hi16=0 | tile_dim0 = 32
  g1[4] = 256;                                       // tile_dim1 = 256 | tile_dim2 = 0
  g1[5] = 1024;                                      // tensor_dim0_stride lo32
  g1[6] = 0;
  g1[7] = 0;
  vi4 z4 = {0, 0, 0, 0};
#if defined(__clang_major__) && (__clang_major__ >= 23)
  vi8 z8 = {0, 0, 0, 0, 0, 0, 0, 0};
  __builtin_amdgcn_tensor_load_to_lds(g0, g1, z4, z4, z8, 0);
#else
  __builtin_amdgcn_tensor_load_to_lds(g0, g1, z4, z4, 0);
#endif
}
#endif

__global__ void __launch_bounds__(256)
qkv_proj_kernel(const float* __restrict__ x,
                const float* __restrict__ Wq, const float* __restrict__ bq,
                const float* __restrict__ Wk, const float* __restrict__ bk,
                const float* __restrict__ Wv, const float* __restrict__ bv,
                _Float16* __restrict__ qout,   // [B,H,S,HD]
                _Float16* __restrict__ kout,   // [B,H,S,HD]
                _Float16* __restrict__ vtout)  // [B,H,HD,S] (transposed for PV B-frags)
{
  __shared__ __align__(128) _Float16 As[GM_BM * LDS_STRIDE];
  __shared__ __align__(128) _Float16 Bs[GM_BN * LDS_STRIDE];

  const int tid  = threadIdx.x;
  const int lane = tid & 31;
  const int wv   = tid >> 5;
  const int g    = lane >> 4;
  const int r    = lane & 15;
  const int m0   = blockIdx.x * GM_BM;
  const int n0   = blockIdx.y * GM_BN;
  const int which = blockIdx.z;

  const float* W    = (which == 0) ? Wq : ((which == 1) ? Wk : Wv);
  const float* bias = (which == 0) ? bq : ((which == 1) ? bk : bv);

  float8_t acc[2][4] = {};

  for (int k0 = 0; k0 < DD; k0 += GM_BK) {
    __syncthreads();
    // Stage A tile 256x32 f32 -> f16: issue ALL 8 global_load_b128 first so
    // LOADcnt waits are staggered, then convert + ds_store. Prefetch next
    // K-step unconditionally (speculative prefetch is dropped on bad xlat).
    float4_t va[8];
#pragma unroll
    for (int i = 0; i < 8; ++i) {
      int f = tid + i * 256;
      int row = f >> 3, c4 = f & 7;
      const float* gp = x + (size_t)(m0 + row) * DD + k0 + c4 * 4;
      va[i] = *(const float4_t*)gp;
      __builtin_prefetch(gp + GM_BK, 0, 3);
    }
    float4_t vw[2];
#pragma unroll
    for (int i = 0; i < 2; ++i) {
      int f = tid + i * 256;
      int row = f >> 3, c4 = f & 7;
      const float* gp = W + (size_t)(n0 + row) * DD + k0 + c4 * 4;
      vw[i] = *(const float4_t*)gp;
      __builtin_prefetch(gp + GM_BK, 0, 3);
    }
#pragma unroll
    for (int i = 0; i < 8; ++i) {
      int f = tid + i * 256;
      int row = f >> 3, c4 = f & 7;
      half4_t hv;
      hv[0] = (_Float16)va[i][0]; hv[1] = (_Float16)va[i][1];
      hv[2] = (_Float16)va[i][2]; hv[3] = (_Float16)va[i][3];
      *(half4_t*)&As[row * LDS_STRIDE + c4 * 4] = hv;
    }
#pragma unroll
    for (int i = 0; i < 2; ++i) {
      int f = tid + i * 256;
      int row = f >> 3, c4 = f & 7;
      half4_t hv;
      hv[0] = (_Float16)vw[i][0]; hv[1] = (_Float16)vw[i][1];
      hv[2] = (_Float16)vw[i][2]; hv[3] = (_Float16)vw[i][3];
      *(half4_t*)&Bs[row * LDS_STRIDE + c4 * 4] = hv;
    }
    __syncthreads();

    // Load ALL fragments first, then issue 8 WMMAs back-to-back.
    const int arow0 = wv * 32 + r;
    const int arow1 = arow0 + 16;
    half16_t a0 = cat16(*(const half8_t*)&As[arow0 * LDS_STRIDE + g * 8],
                        *(const half8_t*)&As[arow0 * LDS_STRIDE + 16 + g * 8]);
    half16_t a1 = cat16(*(const half8_t*)&As[arow1 * LDS_STRIDE + g * 8],
                        *(const half8_t*)&As[arow1 * LDS_STRIDE + 16 + g * 8]);
    half16_t bf[4];
#pragma unroll
    for (int t = 0; t < 4; ++t) {
      const int bcol = t * 16 + r;
      bf[t] = cat16(*(const half8_t*)&Bs[bcol * LDS_STRIDE + g * 8],
                    *(const half8_t*)&Bs[bcol * LDS_STRIDE + 16 + g * 8]);
    }
#pragma unroll
    for (int t = 0; t < 4; ++t) acc[0][t] = wmma_f32_16x16x32_f16(a0, bf[t], acc[0][t]);
#pragma unroll
    for (int t = 0; t < 4; ++t) acc[1][t] = wmma_f32_16x16x32_f16(a1, bf[t], acc[1][t]);
  }

  // Epilogue: C layout -> element (m = g*8+rr, n = r) per VGPR rr
#pragma unroll
  for (int hf = 0; hf < 2; ++hf) {
#pragma unroll
    for (int t = 0; t < 4; ++t) {
      const int n = n0 + t * 16 + r;
      const float bval = bias[n];
      const int h = n >> 6, hd = n & 63;
#pragma unroll
      for (int rr = 0; rr < 8; ++rr) {
        const int m = m0 + wv * 32 + hf * 16 + g * 8 + rr;
        const int b = m >> 11;        // / S
        const int s = m & (SS - 1);
        const _Float16 val = (_Float16)(acc[hf][t][rr] + bval);
        if (which == 2) {
          vtout[(((size_t)b * HH + h) * HDIM + hd) * SS + s] = val;
        } else {
          _Float16* o = (which == 0) ? qout : kout;
          o[(((size_t)b * HH + h) * SS + s) * HDIM + hd] = val;
        }
      }
    }
  }
}

// ---------------------------------------------------------------------------
// Flash attention: one wave per 32-query tile (2 Q-frag pairs, 8 accumulators)
// so every K and V fragment feeds TWO WMMAs: 16 WMMAs per 32-key chunk for
// 16 global b128 loads. Online softmax with shfl_xor row reductions; P repack
// via per-wave LDS tile guarded by s_wait_dscnt.
// ---------------------------------------------------------------------------
__global__ void __launch_bounds__(128)
flash_attn_kernel(const _Float16* __restrict__ Q,   // [B,H,S,HD]
                  const _Float16* __restrict__ Kh,  // [B,H,S,HD]
                  const _Float16* __restrict__ Vt,  // [B,H,HD,S]
                  const unsigned char* __restrict__ pmask, // [B,S] bool
                  _Float16* __restrict__ ctx)       // [B,S,D]
{
  __shared__ _Float16 Plds[4][32 * 32];

  const int tid  = threadIdx.x;
  const int lane = tid & 31;
  const int wv   = tid >> 5;
  const int g    = lane >> 4;
  const int r    = lane & 15;
  const int q0   = (blockIdx.x * 4 + wv) * 32;
  const int bh   = blockIdx.y;
  const int b    = bh >> 4;   // / H
  const int h    = bh & 15;

  const _Float16* Qb = Q  + (size_t)bh * SS * HDIM;
  const _Float16* Kb = Kh + (size_t)bh * SS * HDIM;
  const _Float16* Vb = Vt + (size_t)bh * HDIM * SS;
  const unsigned char* pm = pmask + (size_t)b * SS;
  _Float16* Pw = &Plds[wv][0];

  // Q fragments: [query-half][d-half], kept in registers across the key loop
  half16_t qa[2][2];
#pragma unroll
  for (int qh = 0; qh < 2; ++qh) {
    const _Float16* qrow = Qb + (size_t)(q0 + qh * 16 + r) * HDIM;
    qa[qh][0] = cat16(*(const half8_t*)(qrow + g * 8),
                      *(const half8_t*)(qrow + 16 + g * 8));
    qa[qh][1] = cat16(*(const half8_t*)(qrow + 32 + g * 8),
                      *(const half8_t*)(qrow + 48 + g * 8));
  }

  float8_t acc[2][4] = {};
  float mrow[2][8], lrow[2][8];
#pragma unroll
  for (int qh = 0; qh < 2; ++qh)
#pragma unroll
    for (int i = 0; i < 8; ++i) { mrow[qh][i] = -1e30f; lrow[qh][i] = 0.0f; }

  const float scale = 0.125f;        // 1/sqrt(64)
  const int nchunk = (q0 >> 5) + 1;  // causal: keys <= q0+31 needed

  for (int c = 0; c < nchunk; ++c) {
    const int kc = c << 5;
    __builtin_prefetch(Kb + (size_t)(kc + 32 + r) * HDIM, 0, 3);  // next chunk

    float p[2][2][8];   // [key-tile][query-half][rr]
    float cm[2][8];     // [query-half][rr]
#pragma unroll
    for (int qh = 0; qh < 2; ++qh)
#pragma unroll
      for (int rr = 0; rr < 8; ++rr) cm[qh][rr] = -1e30f;

#pragma unroll
    for (int t = 0; t < 2; ++t) {
      const int k0 = kc + t * 16;
      // B fragment of K^T: lane col = key (k0+r), halves i -> d = g*8+i / 16+g*8+i-8;
      // shared by both query halves (2x reuse).
      const _Float16* krow = Kb + (size_t)(k0 + r) * HDIM;
      const half16_t kb0 = cat16(*(const half8_t*)(krow + g * 8),
                                 *(const half8_t*)(krow + 16 + g * 8));
      const half16_t kb1 = cat16(*(const half8_t*)(krow + 32 + g * 8),
                                 *(const half8_t*)(krow + 48 + g * 8));
      const bool kvalid = pm[k0 + r] != 0;
      const int  kcol   = k0 + r;
#pragma unroll
      for (int qh = 0; qh < 2; ++qh) {
        float8_t s = {};
        s = wmma_f32_16x16x32_f16(qa[qh][0], kb0, s);
        s = wmma_f32_16x16x32_f16(qa[qh][1], kb1, s);
#pragma unroll
        for (int rr = 0; rr < 8; ++rr) {
          const int qidx = q0 + qh * 16 + g * 8 + rr;
          const float sv = (!kvalid || (kcol > qidx)) ? -1e30f : s[rr] * scale;
          p[t][qh][rr] = sv;
          cm[qh][rr] = fmaxf(cm[qh][rr], sv);
        }
      }
    }

    // Row reductions stay inside each 16-lane half (xor masks 1,2,4,8)
#pragma unroll
    for (int qh = 0; qh < 2; ++qh) {
#pragma unroll
      for (int rr = 0; rr < 8; ++rr) {
        float v = cm[qh][rr];
        v = fmaxf(v, __shfl_xor(v, 1));
        v = fmaxf(v, __shfl_xor(v, 2));
        v = fmaxf(v, __shfl_xor(v, 4));
        v = fmaxf(v, __shfl_xor(v, 8));
        const float nm    = fmaxf(mrow[qh][rr], v);
        const float alpha = __expf(mrow[qh][rr] - nm);
        mrow[qh][rr] = nm;
        const float p0 = __expf(p[0][qh][rr] - nm);
        const float p1 = __expf(p[1][qh][rr] - nm);
        p[0][qh][rr] = p0; p[1][qh][rr] = p1;
        float rs = p0 + p1;
        rs += __shfl_xor(rs, 1);
        rs += __shfl_xor(rs, 2);
        rs += __shfl_xor(rs, 4);
        rs += __shfl_xor(rs, 8);
        lrow[qh][rr] = lrow[qh][rr] * alpha + rs;
#pragma unroll
        for (int t = 0; t < 4; ++t) acc[qh][t][rr] *= alpha;
      }
    }

    // Repack P (C-layout: m=qh*16+g*8+rr, n=r) into A-fragment layout via LDS
#pragma unroll
    for (int qh = 0; qh < 2; ++qh) {
#pragma unroll
      for (int rr = 0; rr < 8; ++rr) {
        const int m = qh * 16 + g * 8 + rr;
        Pw[m * 32 + r]      = (_Float16)p[0][qh][rr];
        Pw[m * 32 + 16 + r] = (_Float16)p[1][qh][rr];
      }
    }
    asm volatile("s_wait_dscnt 0" ::: "memory");  // cross-lane LDS RAW within wave

    const half16_t pa0 = cat16(*(const half8_t*)&Pw[r * 32 + g * 8],
                               *(const half8_t*)&Pw[r * 32 + 16 + g * 8]);
    const half16_t pa1 = cat16(*(const half8_t*)&Pw[(16 + r) * 32 + g * 8],
                               *(const half8_t*)&Pw[(16 + r) * 32 + 16 + g * 8]);

    // PV: B fragment from V^T (lane col = d, halves i -> key = kc+g*8+i / kc+16+g*8+i-8),
    // each V fragment shared by both query halves (2x reuse).
    half16_t vbf[4];
#pragma unroll
    for (int t = 0; t < 4; ++t) {
      const _Float16* vrow = Vb + (size_t)(t * 16 + r) * SS + kc;
      vbf[t] = cat16(*(const half8_t*)(vrow + g * 8),
                     *(const half8_t*)(vrow + 16 + g * 8));
    }
#pragma unroll
    for (int t = 0; t < 4; ++t) acc[0][t] = wmma_f32_16x16x32_f16(pa0, vbf[t], acc[0][t]);
#pragma unroll
    for (int t = 0; t < 4; ++t) acc[1][t] = wmma_f32_16x16x32_f16(pa1, vbf[t], acc[1][t]);
  }

  // Epilogue: ctx[b, s, h*64 + d] = acc / l
#pragma unroll
  for (int qh = 0; qh < 2; ++qh) {
#pragma unroll
    for (int rr = 0; rr < 8; ++rr) {
      const float inv = 1.0f / lrow[qh][rr];
      const int   s   = q0 + qh * 16 + g * 8 + rr;
      const size_t base = ((size_t)b * SS + s) * DD + h * HDIM;
#pragma unroll
      for (int t = 0; t < 4; ++t) {
        ctx[base + t * 16 + r] = (_Float16)(acc[qh][t][rr] * inv);
      }
    }
  }
}

// ---------------------------------------------------------------------------
// Output projection: ctx(f16) @ Wo^T + bo -> f32 out. Same tiling as QKV GEMM;
// the f16 A tile is DMA'd into LDS by the Tensor Data Mover (LDS row padding
// applied by the TDM), synchronized via s_wait_tensorcnt + barrier.
// ---------------------------------------------------------------------------
__global__ void __launch_bounds__(256)
out_proj_kernel(const _Float16* __restrict__ A,   // [B*S, D] f16
                const float* __restrict__ W,      // [D, D] f32 (row = out feature)
                const float* __restrict__ bias,
                float* __restrict__ out)          // [B*S, D] f32
{
  __shared__ __align__(128) _Float16 As[GM_BM * LDS_STRIDE];
  __shared__ __align__(128) _Float16 Bs[GM_BN * LDS_STRIDE];

  const int tid  = threadIdx.x;
  const int lane = tid & 31;
  const int wv   = tid >> 5;
  const int g    = lane >> 4;
  const int r    = lane & 15;
  const int m0   = blockIdx.x * GM_BM;
  const int n0   = blockIdx.y * GM_BN;

  float8_t acc[2][4] = {};

  for (int k0 = 0; k0 < DD; k0 += GM_BK) {
    __syncthreads();
#if HAVE_TDM
    if (tid < 32) {  // wave 0 issues the TDM load of the 256x32 f16 A tile
      tdm_load_a_tile(A + (size_t)m0 * DD + k0, (unsigned int)(size_t)&As[0]);
    }
#else
    // Fallback: VGPR staging of A tile 256x32 f16 (1024 b128 / 256 thr = 4 each)
    half8_t ha[4];
#pragma unroll
    for (int i = 0; i < 4; ++i) {
      int f = tid + i * 256;
      int row = f >> 2, c8 = f & 3;
      ha[i] = *(const half8_t*)(A + (size_t)(m0 + row) * DD + k0 + c8 * 8);
    }
#pragma unroll
    for (int i = 0; i < 4; ++i) {
      int f = tid + i * 256;
      int row = f >> 2, c8 = f & 3;
      *(half8_t*)&As[row * LDS_STRIDE + c8 * 8] = ha[i];
    }
#endif
    // Stage W tile 64x32 f32 -> f16 (overlaps with the TDM transfer)
    float4_t vw[2];
#pragma unroll
    for (int i = 0; i < 2; ++i) {
      int f = tid + i * 256;
      int row = f >> 3, c4 = f & 7;
      const float* gp = W + (size_t)(n0 + row) * DD + k0 + c4 * 4;
      vw[i] = *(const float4_t*)gp;
      __builtin_prefetch(gp + GM_BK, 0, 3);
    }
#pragma unroll
    for (int i = 0; i < 2; ++i) {
      int f = tid + i * 256;
      int row = f >> 3, c4 = f & 7;
      half4_t hv;
      hv[0] = (_Float16)vw[i][0]; hv[1] = (_Float16)vw[i][1];
      hv[2] = (_Float16)vw[i][2]; hv[3] = (_Float16)vw[i][3];
      *(half4_t*)&Bs[row * LDS_STRIDE + c4 * 4] = hv;
    }
#if HAVE_TDM
    if (tid < 32) __builtin_amdgcn_s_wait_tensorcnt(0);
#endif
    __syncthreads();

    const int arow0 = wv * 32 + r;
    const int arow1 = arow0 + 16;
    half16_t a0 = cat16(*(const half8_t*)&As[arow0 * LDS_STRIDE + g * 8],
                        *(const half8_t*)&As[arow0 * LDS_STRIDE + 16 + g * 8]);
    half16_t a1 = cat16(*(const half8_t*)&As[arow1 * LDS_STRIDE + g * 8],
                        *(const half8_t*)&As[arow1 * LDS_STRIDE + 16 + g * 8]);
    half16_t bf[4];
#pragma unroll
    for (int t = 0; t < 4; ++t) {
      const int bcol = t * 16 + r;
      bf[t] = cat16(*(const half8_t*)&Bs[bcol * LDS_STRIDE + g * 8],
                    *(const half8_t*)&Bs[bcol * LDS_STRIDE + 16 + g * 8]);
    }
#pragma unroll
    for (int t = 0; t < 4; ++t) acc[0][t] = wmma_f32_16x16x32_f16(a0, bf[t], acc[0][t]);
#pragma unroll
    for (int t = 0; t < 4; ++t) acc[1][t] = wmma_f32_16x16x32_f16(a1, bf[t], acc[1][t]);
  }

#pragma unroll
  for (int hf = 0; hf < 2; ++hf) {
#pragma unroll
    for (int t = 0; t < 4; ++t) {
      const int n = n0 + t * 16 + r;
      const float bval = bias[n];
#pragma unroll
      for (int rr = 0; rr < 8; ++rr) {
        const int m = m0 + wv * 32 + hf * 16 + g * 8 + rr;
        out[(size_t)m * DD + n] = acc[hf][t][rr] + bval;
      }
    }
  }
}

// ---------------------------------------------------------------------------
extern "C" void kernel_launch(void* const* d_in, const int* in_sizes, int n_in,
                              void* d_out, int out_size, void* d_ws, size_t ws_size,
                              hipStream_t stream) {
  const float*         x     = (const float*)d_in[0];
  const unsigned char* pmask = (const unsigned char*)d_in[1];  // jax bool: 1 byte
  const float*         Wq    = (const float*)d_in[2];
  const float*         bq    = (const float*)d_in[3];
  const float*         Wk    = (const float*)d_in[4];
  const float*         bk    = (const float*)d_in[5];
  const float*         Wv    = (const float*)d_in[6];
  const float*         bv    = (const float*)d_in[7];
  const float*         Wo    = (const float*)d_in[8];
  const float*         bo    = (const float*)d_in[9];
  float*               out   = (float*)d_out;

  const size_t elems = (size_t)BB * SS * DD;   // 8,388,608
  _Float16* qws  = (_Float16*)d_ws;            // [B,H,S,HD]   16 MB
  _Float16* kws  = qws + elems;                // [B,H,S,HD]   16 MB
  _Float16* vtws = kws + elems;                // [B,H,HD,S]   16 MB
  _Float16* cws  = vtws + elems;               // [B,S,D]      16 MB

  dim3 gq((BB * SS) / GM_BM, DD / GM_BN, 3);   // 32 x 16 x 3
  qkv_proj_kernel<<<gq, 256, 0, stream>>>(x, Wq, bq, Wk, bk, Wv, bv, qws, kws, vtws);

  dim3 ga(SS / 128, BB * HH);                  // 16 x 64 (4 x 32-query tiles / block)
  flash_attn_kernel<<<ga, 128, 0, stream>>>(qws, kws, vtws, pmask, cws);

  dim3 go((BB * SS) / GM_BM, DD / GM_BN);      // 32 x 16
  out_proj_kernel<<<go, 256, 0, stream>>>(cws, Wo, bo, out);
}
